// KappaGCN_70660801954418
// MI455X (gfx1250) — compile-verified, matrix operand order checked by reading, not run on hardware
//
#include <hip/hip_runtime.h>

// ---------------------------------------------------------------------------
// CDNA5 (gfx1250) kappa-GCN: bf16 WMMA for all GEMMs (f32 accumulate),
// fp32 VALU for the hyperbolic nonlinearities.
//
// Roofline: A_hat (268 MB fp32) can't live in the 192 MB L2, so both big
// GEMMs stream it from HBM. Converting A to bf16 once makes the heavy
// 8192x8192x288 GEMM memory-bound (~134 MB/pass at 23.3 TB/s) on the
// v_wmma_f32_16x16x32_bf16 path. All B operands are stored
// K-contiguous (transposed) so every WMMA fragment is two b128 loads/lane.
//
// Workspace layout (bytes):
//   Ab : bf16 copy of A_hat              8192*8192*2 = 134,217,728
//   Wb : bf16 copy of W                    256*256*2 =     131,072
//   Bt : bf16 [gamma*XW | gamma-1 | 1 | 0] transposed (288 x 8192) = 4,718,592
//   Lgt: bf16 logits transposed (16 x 8192)           =     262,144
//   Cst: per-class constants                          =       1,024
// Total ~139.3 MB.
// ---------------------------------------------------------------------------

typedef __attribute__((ext_vector_type(16))) __bf16 v16bf;
typedef __attribute__((ext_vector_type(8)))  float  v8f;
typedef __attribute__((ext_vector_type(4)))  int    v4i;
typedef unsigned short u16;
typedef unsigned int   u32;
typedef unsigned long long u64;

union ABFrag { v16bf v; u16 u[16]; u32 w[8]; };

// --- optional async-to-LDS path (gfx1250 GLOBAL_LOAD_ASYNC_TO_LDS_B128) ----
#if defined(__has_builtin)
#if __has_builtin(__builtin_amdgcn_global_load_async_to_lds_b128)
#define USE_ASYNC_LDS 1
#endif
#if __has_builtin(__builtin_amdgcn_s_wait_asynccnt)
#define HAVE_WAIT_ASYNC_BUILTIN 1
#endif
#endif

#if defined(USE_ASYNC_LDS)
static __device__ __forceinline__ void async_copy_b128(const u16* gsrc, u16* ldst) {
  // Builtin signature (from probe): (int4* src, int4* dst, imm offset, imm cpol)
  __builtin_amdgcn_global_load_async_to_lds_b128((v4i*)gsrc, (v4i*)ldst, 0, 0);
}

static __device__ __forceinline__ void wait_async0() {
#if defined(HAVE_WAIT_ASYNC_BUILTIN)
  __builtin_amdgcn_s_wait_asynccnt(0);
#else
  asm volatile("s_wait_asynccnt 0x0" ::: "memory");
#endif
}
#endif
// ---------------------------------------------------------------------------

__device__ __forceinline__ u16 f2bf(float f) {
  union { float f; u32 u; } x{f};
  u32 r = x.u + 0x7FFFu + ((x.u >> 16) & 1u);   // round-to-nearest-even
  return (u16)(r >> 16);
}

__device__ __forceinline__ float artanh_c(float x) {
  // reference: arctanh(clip(x, -1+1e-7, 1-1e-7))
  x = fminf(fmaxf(x, -1.f + 1e-7f), 1.f - 1e-7f);
  return 0.5f * logf((1.f + x) / (1.f - x));
}

__device__ __forceinline__ v8f zero8() {
  v8f z = {0.f, 0.f, 0.f, 0.f, 0.f, 0.f, 0.f, 0.f};
  return z;
}

// K-offset pattern for 16-bit operand fragments (ISA 7.12.2, 16-bit A 16x32):
// VGPR pair p holds K = {2p+8h, 2p+1+8h} for p<4, {16+2(p-4)+8h, +1} for p>=4
__device__ __forceinline__ int kpat(int p, int h) {
  return (p < 4) ? (2 * p + 8 * h) : (16 + 2 * (p - 4) + 8 * h);
}

// A fragment: 16x32 bf16 tile from row-major src (K contiguous)
__device__ __forceinline__ v16bf load_afrag(const u16* A, int lda, int row0, int k0) {
  const int lane = threadIdx.x & 31;
  const int r = lane & 15, h = lane >> 4;
  ABFrag f;
  const u16* base = A + (row0 + r) * lda + k0;
#pragma unroll
  for (int p = 0; p < 8; ++p) {
    f.w[p] = *(const u32*)(base + kpat(p, h));   // merges into 2x b128
  }
  return f.v;
}

// B fragment: 32x16 tile from column-major (N x K, K contiguous) storage:
// element (k, c) at B[(c0+c)*ld + k]  -> same contiguous pattern as A.
__device__ __forceinline__ v16bf load_bfrag_nk(const u16* B, int ld, int k0, int c0) {
  const int lane = threadIdx.x & 31;
  const int c = lane & 15, h = lane >> 4;
  ABFrag f;
  const u16* base = B + (c0 + c) * ld + k0;
#pragma unroll
  for (int p = 0; p < 8; ++p) {
    f.w[p] = *(const u32*)(base + kpat(p, h));
  }
  return f.v;
}

__device__ __forceinline__ v8f wmma_bf16(v16bf a, v16bf b, v8f c) {
  return __builtin_amdgcn_wmma_f32_16x16x32_bf16(false, a, false, b,
                                                 (short)0, c, false, false);
}

// ---------------------------------------------------------------------------
// K1: fp32 -> bf16 bulk convert (A_hat and W). Memory bound, float4 loads.
// ---------------------------------------------------------------------------
__global__ void k_f32_to_bf16(const float* __restrict__ src,
                              u16* __restrict__ dst, long long n4) {
  long long i = (long long)blockIdx.x * blockDim.x + threadIdx.x;
  const long long stride = (long long)gridDim.x * blockDim.x;
  for (; i < n4; i += stride) {
    const float4 v = ((const float4*)src)[i];
    u64 pk = (u64)f2bf(v.x) | ((u64)f2bf(v.y) << 16) |
             ((u64)f2bf(v.z) << 32) | ((u64)f2bf(v.w) << 48);
    ((u64*)dst)[i] = pk;
  }
}

// ---------------------------------------------------------------------------
// K0: per-class constants: |p_k|^2, <p_k, wl_k>, |wl_k|, lambda(p_k)
// ---------------------------------------------------------------------------
__global__ void k_class_consts(const float* __restrict__ Wl,
                               const float* __restrict__ Pk,
                               float* __restrict__ C) {
  const int k = threadIdx.x;
  if (k >= 16) return;
  float x2 = 0.f, pw = 0.f, an2 = 0.f;
  for (int d = 0; d < 256; ++d) {
    const float p = Pk[k * 256 + d];
    const float wv = Wl[d * 16 + k];
    x2 += p * p;  pw += p * wv;  an2 += wv * wv;
  }
  C[k]      = x2;
  C[16 + k] = pw;
  C[32 + k] = fmaxf(sqrtf(an2), 1e-15f);
  C[48 + k] = 2.f / fmaxf(1.f - x2, 1e-10f);    // lambda_x(p_k), K=-1
}

// ---------------------------------------------------------------------------
// K2: fused mobius_matvec + lambda.  Writes Bt TRANSPOSED (288 x 8192):
//   Bt[c][row] = gamma*XW  (c<256),  Bt[256][row]=gamma-1,  Bt[257][row]=1.
// 512 threads = 16 waves; wave w owns output cols [16w,16w+16).
// Each lane holds 8 consecutive rows of one column -> single b128 store.
// ---------------------------------------------------------------------------
__global__ __launch_bounds__(512)
void k_mobius_matvec(const float* __restrict__ X, const u16* __restrict__ Wb,
                     u16* __restrict__ Bt) {
  __shared__ __align__(16) u16 s_x[16 * 256];   // X row-block as bf16
  __shared__ float s_xn2[16], s_mx2[16], s_bs[16], s_gm1[16];
  const int t = threadIdx.x;
  const int row0 = blockIdx.x * 16;
  if (t < 16) { s_xn2[t] = 0.f; s_mx2[t] = 0.f; }
  __syncthreads();
  for (int i = t; i < 16 * 256; i += 512) {
    const int r = i >> 8;
    const float x = X[(row0 + r) * 256 + (i & 255)];
    s_x[i] = f2bf(x);
    atomicAdd(&s_xn2[r], x * x);                // |x|^2 per row
  }
  __syncthreads();

  const int w = t >> 5, lane = t & 31, hh = lane >> 4, cc = lane & 15;
  v8f acc = zero8();
#pragma unroll
  for (int k0 = 0; k0 < 256; k0 += 32) {        // mx = X @ W^T
    acc = wmma_bf16(load_afrag(s_x, 256, 0, k0),
                    load_bfrag_nk(Wb, 256, k0, w * 16), acc);
  }
#pragma unroll
  for (int v = 0; v < 8; ++v)                   // |mx|^2 per row
    atomicAdd(&s_mx2[v + 8 * hh], acc[v] * acc[v]);
  __syncthreads();

  if (t < 16) {
    const float xn  = fmaxf(sqrtf(fmaxf(s_xn2[t], 0.f)), 1e-15f);
    const float mxn = fmaxf(sqrtf(fmaxf(s_mx2[t], 0.f)), 1e-15f);
    const float sc  = tanhf((mxn / xn) * artanh_c(xn)) / mxn; // XW = sc*mx
    const float xwn = sc * mxn;                                // |XW|
    const float gam = 2.f / fmaxf(1.f - xwn * xwn, 1e-10f);    // lambda_x
    s_bs[t]  = gam * sc;
    s_gm1[t] = gam - 1.f;
  }
  __syncthreads();

  {                                             // transposed b128 store
    const int col = w * 16 + cc;
    const int rbase = 8 * hh;                   // rows rbase..rbase+7
    u32 pk[4];
#pragma unroll
    for (int q = 0; q < 4; ++q) {
      const u16 lo = f2bf(s_bs[rbase + 2 * q]     * acc[2 * q]);
      const u16 hi = f2bf(s_bs[rbase + 2 * q + 1] * acc[2 * q + 1]);
      pk[q] = (u32)lo | ((u32)hi << 16);
    }
    u32* dst = (u32*)&Bt[col * 8192 + row0 + rbase];  // 16B aligned
    *(uint4*)dst = make_uint4(pk[0], pk[1], pk[2], pk[3]);
  }
  if (t < 16) {
    Bt[256 * 8192 + row0 + t] = f2bf(s_gm1[t]);
    Bt[257 * 8192 + row0 + t] = f2bf(1.f);
    for (int c = 258; c < 288; ++c) Bt[c * 8192 + row0 + t] = 0;
  }
}

// ---------------------------------------------------------------------------
// K3: heavy GEMM C = Ab(8192x8192) @ B(8192x288, stored transposed) with the
// full manifold epilogue fused (gyromidpoint -> mobius scalar muls ->
// logmap0 -> relu -> expmap0 -> hyperbolic logits).
// 576 threads = 18 waves = 18 N-tiles x 16 rows. A slab staged through LDS
// in 128-wide chunks (async-to-LDS when available), streamed from HBM once.
// ---------------------------------------------------------------------------
__global__ __launch_bounds__(576)
void k_main_gemm(const u16* __restrict__ Ab, const u16* __restrict__ Bt,
                 const float* __restrict__ Wl, const float* __restrict__ Pk,
                 const float* __restrict__ Cst, u16* __restrict__ Lgt) {
  __shared__ __align__(16) u16 s_a[16 * 128];   // staged A chunk (4 KB)
  __shared__ float s_h[16 * 256];               // H1 row-block (16 KB)
  __shared__ float s_den[16], s_rs[16], s_tm2[16], s_vn2[16];
  __shared__ float s_ul[16], s_hs[16], s_y2[16];

  const int t = threadIdx.x;
  const int row0 = blockIdx.x * 16;
  const int w = t >> 5, lane = t & 31, hh = lane >> 4, cc = lane & 15;
  if (t < 16) { s_tm2[t] = 0.f; s_vn2[t] = 0.f; }

  v8f acc = zero8();
  for (int kb = 0; kb < 8192; kb += 128) {
    __syncthreads();                            // s_a consumers done
#if defined(USE_ASYNC_LDS)
    if (t < 256) {                              // 16 rows x 16 b128 each
      const int r = t >> 4, q = t & 15;
      async_copy_b128(&Ab[(long long)(row0 + r) * 8192 + kb + q * 8],
                      &s_a[r * 128 + q * 8]);
      wait_async0();
    }
#else
    if (t < 512) {                              // 16 rows x 32 b64 each
      const int r = t >> 5, q = t & 31;
      *(uint2*)&s_a[r * 128 + q * 4] =
          *(const uint2*)&Ab[(long long)(row0 + r) * 8192 + kb + q * 4];
      if (kb + 128 < 8192)                      // prefetch next A chunk
        __builtin_prefetch(&Ab[(long long)(row0 + r) * 8192 + kb + 128 + q * 4], 0, 3);
    }
#endif
    __syncthreads();
#pragma unroll
    for (int kk = 0; kk < 128; kk += 32) {
      acc = wmma_bf16(load_afrag(s_a, 128, 0, kk),
                      load_bfrag_nk(Bt, 8192, kb + kk, w * 16), acc);
    }
  }
  __syncthreads();

  // wave 16 holds cols 256 (denom = A@(gamma-1)) and 257 (rowsum = A@1)
  if (w == 16 && cc < 2) {
#pragma unroll
    for (int v = 0; v < 8; ++v) {
      const int r = v + 8 * hh;
      if (cc == 0) s_den[r] = acc[v]; else s_rs[r] = acc[v];
    }
  }
  __syncthreads();

  float tm[8] = {0.f, 0.f, 0.f, 0.f, 0.f, 0.f, 0.f, 0.f};
  if (w < 16) {
#pragma unroll
    for (int v = 0; v < 8; ++v) {
      const int r = v + 8 * hh;
      float den = s_den[r];
      den = (fabsf(den) < 1e-10f) ? 1e-10f : den;
      tm[v] = acc[v] / den;                     // two_mean
      atomicAdd(&s_tm2[r], tm[v] * tm[v]);
    }
  }
  __syncthreads();

  if (t < 16) {
    // logmap0(msm(rowsum, msm(0.5, tm))) collapses to (arg/|tm|)*tm with
    // arg = rowsum * 0.5 * artanh(clip(|tm|))  (norm terms cancel exactly)
    const float tmn = fmaxf(sqrtf(fmaxf(s_tm2[t], 0.f)), 1e-15f);
    const float arg = s_rs[t] * 0.5f * artanh_c(tmn);
    s_ul[t] = arg / tmn;
  }
  __syncthreads();

  float ve[8] = {0.f, 0.f, 0.f, 0.f, 0.f, 0.f, 0.f, 0.f};
  if (w < 16) {
#pragma unroll
    for (int v = 0; v < 8; ++v) {
      const int r = v + 8 * hh;
      ve[v] = fmaxf(s_ul[r] * tm[v], 0.f);      // relu(logmap0(AXW))
      atomicAdd(&s_vn2[r], ve[v] * ve[v]);
    }
  }
  __syncthreads();

  if (t < 16) {                                 // expmap0 scalars
    const float vn = fmaxf(sqrtf(fmaxf(s_vn2[t], 0.f)), 1e-15f);
    const float th = tanhf(vn);
    s_hs[t] = th / vn;
    s_y2[t] = th * th;                          // |H1|^2
  }
  __syncthreads();

  if (w < 16) {
#pragma unroll
    for (int v = 0; v < 8; ++v)
      s_h[(v + 8 * hh) * 256 + w * 16 + cc] = s_hs[v + 8 * hh] * ve[v];
  }
  __syncthreads();

  // hyperbolic logits: one (row, class) pair per thread, d-loop in registers
  if (t < 256) {
    const int r = t >> 4, k = t & 15;
    float ph = 0.f, hw = 0.f;
    for (int d = 0; d < 256; ++d) {
      const float h = s_h[r * 256 + d];
      ph += Pk[k * 256 + d] * h;                // <p_k, h>
      hw += Wl[d * 16 + k] * h;                 // <wl_k, h>
    }
    const float x2 = Cst[k], pw = Cst[16 + k], an = Cst[32 + k], lam = Cst[48 + k];
    const float y2 = s_y2[r];
    const float c1 = 1.f - 2.f * ph + y2;       // mobius_add(-p, h), K=-1
    const float c2 = 1.f - x2;
    float den = fmaxf(1.f - 2.f * ph + x2 * y2, 1e-15f);
    const float zn2 = (c1 * c1 * x2 - 2.f * c1 * c2 * ph + c2 * c2 * y2) / (den * den);
    const float za  = (-c1 * pw + c2 * hw) / den;
    const float dist = asinhf(2.f * za / (fmaxf(1.f - zn2, 1e-10f) * an));
    Lgt[k * 8192 + row0 + r] = f2bf(lam * an * dist);   // transposed store
  }
}

// ---------------------------------------------------------------------------
// K4: out = A_hat @ logits (8192x8192x16). Split-K over 8 waves + LDS reduce.
// Logits stored transposed (16 x 8192) -> contiguous B fragments.
// ---------------------------------------------------------------------------
__global__ __launch_bounds__(256)
void k_out_gemm(const u16* __restrict__ Ab, const u16* __restrict__ Lgt,
                float* __restrict__ out) {
  __shared__ float s_c[16 * 16];
  const int t = threadIdx.x;
  const int row0 = blockIdx.x * 16;
  const int w = t >> 5, lane = t & 31, hh = lane >> 4, cc = lane & 15;
  s_c[t] = 0.f;
  __syncthreads();
  v8f acc = zero8();
  const int kbeg = w * 1024;
  for (int kb = kbeg; kb < kbeg + 1024; kb += 32) {
    acc = wmma_bf16(load_afrag(Ab, 8192, row0, kb),
                    load_bfrag_nk(Lgt, 8192, kb, 0), acc);
  }
#pragma unroll
  for (int v = 0; v < 8; ++v)
    atomicAdd(&s_c[(v + 8 * hh) * 16 + cc], acc[v]);
  __syncthreads();
  out[(long long)row0 * 16 + t] = s_c[t];
}

// ---------------------------------------------------------------------------
extern "C" void kernel_launch(void* const* d_in, const int* in_sizes, int n_in,
                              void* d_out, int out_size, void* d_ws, size_t ws_size,
                              hipStream_t stream) {
  (void)in_sizes; (void)n_in; (void)out_size; (void)ws_size;
  const float* X  = (const float*)d_in[0];   // 8192 x 256
  const float* A  = (const float*)d_in[1];   // 8192 x 8192
  const float* W  = (const float*)d_in[2];   // 256 x 256
  const float* Wl = (const float*)d_in[3];   // 256 x 16
  const float* Pk = (const float*)d_in[4];   // 16 x 256
  float* out = (float*)d_out;                // 8192 x 16

  char* p = (char*)d_ws;
  u16* Ab  = (u16*)p;  p += (size_t)8192 * 8192 * 2;
  u16* Wb  = (u16*)p;  p += (size_t)256 * 256 * 2;
  u16* Bt  = (u16*)p;  p += (size_t)288 * 8192 * 2;
  u16* Lgt = (u16*)p;  p += (size_t)16 * 8192 * 2;
  float* Cst = (float*)p;

  k_f32_to_bf16<<<65536, 256, 0, stream>>>(A, Ab, (long long)8192 * 8192 / 4);
  k_f32_to_bf16<<<64, 256, 0, stream>>>(W, Wb, (long long)256 * 256 / 4);
  k_class_consts<<<1, 16, 0, stream>>>(Wl, Pk, Cst);
  k_mobius_matvec<<<512, 512, 0, stream>>>(X, Wb, Bt);
  k_main_gemm<<<512, 576, 0, stream>>>(Ab, Bt, Wl, Pk, Cst, Lgt);
  k_out_gemm<<<512, 256, 0, stream>>>(Ab, Lgt, out);
}
